// Sequence_85787676770749
// MI455X (gfx1250) — matrix operand order repaired and checked by reading, not compiled, not run on hardware
//
#include <hip/hip_runtime.h>

// ---------------------------------------------------------------------------
// 2-layer LSTM (H=512, B=256, T=1024, pred_len=64) persistent-kernel solver
// for MI455X (gfx1250, wave32, WMMA bf16 16x16x32, async global->LDS DMA).
// ---------------------------------------------------------------------------

#define H        512
#define BATCH    256
#define TSEQ     1024
#define PRED     64
#define NSTEPS   (TSEQ + PRED - 1)   // 1087
#define NBLOCKS  64                  // persistent grid
#define NTHREADS 256                 // 8 waves

typedef float    v8f   __attribute__((ext_vector_type(8)));
typedef __bf16   v16bf __attribute__((ext_vector_type(16)));
typedef unsigned int u32x4 __attribute__((ext_vector_type(4)));

union BF16x16 { u32x4 q[2]; v16bf v; };

// ---------------- workspace layout (bytes) ----------------
#define SZ_C      (BATCH * H * 4)        // 512 KB  (f32 cell state)
#define SZ_HBF    (BATCH * H * 2)        // 256 KB  (bf16 hidden, x2 ping-pong)
#define OFF_C1    0
#define OFF_C2    (OFF_C1 + SZ_C)
#define OFF_H1BF  (OFF_C2 + SZ_C)
#define OFF_H2BF  (OFF_H1BF + 2 * SZ_HBF)
#define OFF_OBUF  (OFF_H2BF + 2 * SZ_HBF)
#define OFF_BAR   (OFF_OBUF + BATCH * 4)
#define ZERO_BYTES (OFF_BAR + 256)       // state region zeroed every launch
#define SZ_WBF    (4 * H * H * 2)        // 2 MB per weight in bf16
#define OFF_W1    ZERO_BYTES
#define OFF_W2    (OFF_W1 + SZ_WBF)
#define OFF_W3    (OFF_W2 + SZ_WBF)
#define OFF_WIH1  (OFF_W3 + SZ_WBF)      // 2048 f32
#define OFF_B1    (OFF_WIH1 + 4 * H * 4)
#define OFF_B2    (OFF_B1 + 4 * H * 4)
#define OFF_WLIN  (OFF_B2 + 4 * H * 4)   // 513 f32 (Wlin + blin)
#define WS_NEED   (OFF_WLIN + 4096)

// ---------------- math helpers ----------------
__device__ __forceinline__ float sigm(float x) { return 1.0f / (1.0f + __expf(-x)); }
__device__ __forceinline__ float tanhf_(float x) {
  float e = __expf(-2.0f * x);
  return (1.0f - e) / (1.0f + e);
}

// async global -> LDS 16-byte copy (CDNA5 async path, ASYNCcnt-tracked)
__device__ __forceinline__ void async_g2l_b128(unsigned lds_off, const void* gaddr) {
  asm volatile("global_load_async_to_lds_b128 %0, %1, off"
               :: "v"(lds_off), "v"((unsigned long long)(uintptr_t)gaddr)
               : "memory");
}
__device__ __forceinline__ void wait_async0() {
  asm volatile("s_wait_asynccnt 0x0" ::: "memory");
}

// ---------------- grid-wide barrier (persistent kernel) ----------------
__device__ __forceinline__ void gbar(unsigned* bar) {
  __syncthreads();
  if (threadIdx.x == 0) {
    unsigned* cnt = bar;
    unsigned* gen = bar + 1;
    __threadfence();
    unsigned g0 = __hip_atomic_load(gen, __ATOMIC_ACQUIRE, __HIP_MEMORY_SCOPE_AGENT);
    unsigned a  = __hip_atomic_fetch_add(cnt, 1u, __ATOMIC_ACQ_REL, __HIP_MEMORY_SCOPE_AGENT);
    if (a == (unsigned)(NBLOCKS - 1)) {
      __hip_atomic_store(cnt, 0u, __ATOMIC_RELAXED, __HIP_MEMORY_SCOPE_AGENT);
      __hip_atomic_fetch_add(gen, 1u, __ATOMIC_ACQ_REL, __HIP_MEMORY_SCOPE_AGENT);
    } else {
      while (__hip_atomic_load(gen, __ATOMIC_ACQUIRE, __HIP_MEMORY_SCOPE_AGENT) == g0)
        __builtin_amdgcn_s_sleep(2);
    }
  }
  __syncthreads();
}

// ---------------------------------------------------------------------------
// One 16-kstep GEMM chain: acc[f] += A(16x512 slice) * W^T tile columns
// {f*512 + g*16 .. +15}, f = gate 0..3.  A: bf16 row-major (BATCH x H).
// W: bf16 row-major (4H x H).  B-fragments DMA'd global->LDS asynchronously
// into a double-buffered 2x4KB LDS area shared by the block's 8 waves.
// One block barrier per k-step; fully unrolled so accumulators stay pinned.
// ---------------------------------------------------------------------------
__device__ __forceinline__ void gemm16(const __bf16* __restrict__ A,
                                       const __bf16* __restrict__ W,
                                       int g, int mtile, int tid,
                                       v8f acc[4], unsigned char* ldsB) {
  const int lane  = tid & 31;
  const int nloc  = lane & 15;
  const int halfk = lane >> 4;

  // cooperative fill mapping: 256 threads x 16B = 4KB (4 frags x 1KB)
  const int f    = tid >> 6;        // gate / fragment 0..3
  const int r    = tid & 63;
  const int frow = r >> 2;          // 0..15  (n within tile)
  const int q    = r & 3;
  const int fhk  = q >> 1;          // K half (lanes 0-15 vs 16-31)
  const int fh16 = q & 1;           // which 16B chunk of the lane's 32B
  const int fn   = f * H + g * 16 + frow;          // global gate row (4H x H)
  const int fL   = fhk * 16 + frow;                // destination lane slot
  const unsigned ldsd =
      (unsigned)(uintptr_t)(ldsB + f * 1024 + fL * 32 + fh16 * 16);
  const __bf16* fsrc = W + (size_t)fn * H + fhk * 16 + fh16 * 8;

  // A fragment source: row = mtile*16 + nloc, two 16B chunks per kstep
  const __bf16* arow = A + (size_t)(mtile * 16 + nloc) * H + halfk * 16;

  // prologue: DMA k-step 0 into buffer 0
  async_g2l_b128(ldsd, fsrc);
  wait_async0();
  __syncthreads();

#pragma unroll
  for (int ks = 0; ks < 16; ++ks) {
    // kick off DMA of the next k-step into the other buffer (overlaps WMMA)
    if (ks < 15)
      async_g2l_b128(ldsd + ((unsigned)(ks + 1) & 1u) * 4096u,
                     fsrc + (ks + 1) * 32);

    BF16x16 af;
    af.q[0] = *(const u32x4*)(arow + ks * 32);
    af.q[1] = *(const u32x4*)(arow + ks * 32 + 8);

    const unsigned char* buf = ldsB + (ks & 1) * 4096;
#pragma unroll
    for (int ff = 0; ff < 4; ++ff) {
      BF16x16 bf_;
      const u32x4* bp = (const u32x4*)(buf + ff * 1024 + lane * 32);
      bf_.q[0] = bp[0];
      bf_.q[1] = bp[1];
      acc[ff] = __builtin_amdgcn_wmma_f32_16x16x32_bf16(
          false, af.v, false, bf_.v, (short)0, acc[ff], false, false);
    }

    if (ks < 15) wait_async0();   // my DMA for ks+1 landed
    __syncthreads();              // everyone's DMA landed / reads drained
  }
}

// ---------------------------------------------------------------------------
// prep kernels
// ---------------------------------------------------------------------------
__global__ void prep_weights(const float* __restrict__ whh1,
                             const float* __restrict__ wih2,
                             const float* __restrict__ whh2,
                             __bf16* __restrict__ w1, __bf16* __restrict__ w2,
                             __bf16* __restrict__ w3) {
  int i = blockIdx.x * blockDim.x + threadIdx.x;   // 0 .. 4H*H-1 (exact grid)
  w1[i] = (__bf16)whh1[i];
  w2[i] = (__bf16)wih2[i];
  w3[i] = (__bf16)whh2[i];
}

__global__ void prep_small(const float* __restrict__ wih1,
                           const float* __restrict__ bih1,
                           const float* __restrict__ bhh1,
                           const float* __restrict__ bih2,
                           const float* __restrict__ bhh2,
                           const float* __restrict__ Wlin,
                           const float* __restrict__ blin,
                           float* __restrict__ wih1f, float* __restrict__ b1,
                           float* __restrict__ b2, float* __restrict__ wl) {
  int i = blockIdx.x * blockDim.x + threadIdx.x;   // 0..2047
  if (i < 4 * H) {
    wih1f[i] = wih1[i];
    b1[i]    = bih1[i] + bhh1[i];
    b2[i]    = bih2[i] + bhh2[i];
  }
  if (i < H)  wl[i]   = Wlin[i];
  if (i == 0) wl[H]   = blin[0];
}

// ---------------------------------------------------------------------------
// persistent LSTM kernel
// ---------------------------------------------------------------------------
__global__ __launch_bounds__(NTHREADS) void lstm_persist(
    const float* __restrict__ input,        // (B, T) row-major
    const __bf16* __restrict__ w1,          // Whh1 bf16 (4H x H)
    const __bf16* __restrict__ w2,          // Wih2 bf16
    const __bf16* __restrict__ w3,          // Whh2 bf16
    const float* __restrict__ wih1f,        // 4H
    const float* __restrict__ b1c,          // bih1+bhh1
    const float* __restrict__ b2c,          // bih2+bhh2
    const float* __restrict__ wl,           // Wlin (H) + blin at [H]
    float* __restrict__ c1, float* __restrict__ c2,
    __bf16* __restrict__ h1bf, __bf16* __restrict__ h2bf,  // 2x ping-pong each
    float* __restrict__ obuf, unsigned* __restrict__ bar,
    float* __restrict__ out) {
  __shared__ __align__(16) unsigned char ldsB[8192];   // 2 x 4KB double buffer

  const int tid   = threadIdx.x;
  const int lane  = tid & 31;
  const int wave  = tid >> 5;
  const int g     = blockIdx.x >> 1;        // column group 0..31
  const int mhalf = blockIdx.x & 1;
  const int mtile = mhalf * 8 + wave;       // 0..15
  const int nloc  = lane & 15;
  const int halfk = lane >> 4;
  const int j     = g * 16 + nloc;          // hidden index of this lane's col
  const int bbase = mtile * 16 + halfk * 8; // epilogue batch-row base

  // warm the WGP-level cache with this block's weight columns
  {
    const size_t rowoff = (size_t)(g * 16) * H;
    __builtin_prefetch(w1 + rowoff, 0, 3);
    __builtin_prefetch(w2 + rowoff, 0, 3);
    __builtin_prefetch(w3 + rowoff, 0, 3);
  }

  // step-invariant per-lane gate weights / biases
  const float wi_i = wih1f[j],           wi_f = wih1f[H + j];
  const float wi_g = wih1f[2 * H + j],   wi_o = wih1f[3 * H + j];
  const float b1i = b1c[j],  b1f = b1c[H + j],  b1g = b1c[2 * H + j],  b1o = b1c[3 * H + j];
  const float b2i = b2c[j],  b2f = b2c[H + j],  b2g = b2c[2 * H + j],  b2o = b2c[3 * H + j];

  const int HB = BATCH * H;

#pragma unroll 1
  for (int step = 0; step < NSTEPS; ++step) {
    const int pv = step & 1, cu = pv ^ 1;
    const __bf16* h1p = h1bf + pv * HB;
    __bf16*       h1c = h1bf + cu * HB;
    const __bf16* h2p = h2bf + pv * HB;
    __bf16*       h2c = h2bf + cu * HB;
    const bool isar   = (step >= TSEQ);

    // ---- layer 1 GEMM: gates1 = h1_prev @ Whh1^T ----
    v8f acc[4];
    acc[0] = (v8f){0,0,0,0,0,0,0,0}; acc[1] = acc[0]; acc[2] = acc[0]; acc[3] = acc[0];
    gemm16(h1p, w1, g, mtile, tid, acc, ldsB);

    // ---- layer 1 epilogue: + x*Wih1 + bias, activations, c1/h1 update ----
#pragma unroll
    for (int v = 0; v < 8; ++v) {
      const int b  = bbase + v;
      const float xb = isar ? obuf[b] : input[(size_t)b * TSEQ + step];
      const float gi = sigm (acc[0][v] + xb * wi_i + b1i);
      const float gf = sigm (acc[1][v] + xb * wi_f + b1f);
      const float gg = tanhf_(acc[2][v] + xb * wi_g + b1g);
      const float go = sigm (acc[3][v] + xb * wi_o + b1o);
      const int idx  = b * H + j;
      const float cn = gf * c1[idx] + gi * gg;
      c1[idx]  = cn;
      h1c[idx] = (__bf16)(go * tanhf_(cn));
    }

    // ---- layer 2 partial GEMM (h2_prev @ Whh2^T) before the barrier:
    //      depends only on last step's state, hides barrier skew ----
    v8f acc2[4];
    acc2[0] = (v8f){0,0,0,0,0,0,0,0}; acc2[1] = acc2[0]; acc2[2] = acc2[0]; acc2[3] = acc2[0];
    gemm16(h2p, w3, g, mtile, tid, acc2, ldsB);

    gbar(bar);   // h1_cur visible grid-wide

    // ---- layer 2 remaining GEMM: + h1_cur @ Wih2^T ----
    gemm16(h1c, w2, g, mtile, tid, acc2, ldsB);

    // ---- layer 2 epilogue ----
#pragma unroll
    for (int v = 0; v < 8; ++v) {
      const int b  = bbase + v;
      const float gi = sigm (acc2[0][v] + b2i);
      const float gf = sigm (acc2[1][v] + b2f);
      const float gg = tanhf_(acc2[2][v] + b2g);
      const float go = sigm (acc2[3][v] + b2o);
      const int idx  = b * H + j;
      const float cn = gf * c2[idx] + gi * gg;
      c2[idx]  = cn;
      h2c[idx] = (__bf16)(go * tanhf_(cn));
    }

    gbar(bar);   // h2_cur visible grid-wide

    // ---- output projection (only on prediction-producing steps) ----
    if (step >= TSEQ - 1) {
      const int wid = blockIdx.x * 8 + wave;   // 0..511; first 64 work
      if (wid < 64) {
        for (int rr = 0; rr < 4; ++rr) {
          const int b = wid * 4 + rr;
          float s = 0.0f;
          for (int jj = lane; jj < H; jj += 32)
            s += (float)h2c[b * H + jj] * wl[jj];
#pragma unroll
          for (int off = 16; off > 0; off >>= 1)
            s += __shfl_xor(s, off, 32);
          if (lane == 0) {
            const float o = s + wl[H];
            obuf[b] = o;
            if (b == BATCH - 1) out[step - (TSEQ - 1)] = o;
          }
        }
      }
      gbar(bar);  // obuf ready for next step's layer-1 epilogue
    }
  }
}

// ---------------------------------------------------------------------------
// host entry
// ---------------------------------------------------------------------------
extern "C" void kernel_launch(void* const* d_in, const int* in_sizes, int n_in,
                              void* d_out, int out_size, void* d_ws, size_t ws_size,
                              hipStream_t stream) {
  (void)in_sizes; (void)n_in; (void)out_size;
  if (ws_size < (size_t)WS_NEED) return;

  const float* input = (const float*)d_in[0];
  // d_in[1] = pred_len (fixed 64)
  const float* Wih1 = (const float*)d_in[2];
  const float* Whh1 = (const float*)d_in[3];
  const float* bih1 = (const float*)d_in[4];
  const float* bhh1 = (const float*)d_in[5];
  const float* Wih2 = (const float*)d_in[6];
  const float* Whh2 = (const float*)d_in[7];
  const float* bih2 = (const float*)d_in[8];
  const float* bhh2 = (const float*)d_in[9];
  const float* Wlin = (const float*)d_in[10];
  const float* blin = (const float*)d_in[11];

  char* ws = (char*)d_ws;
  float*    c1    = (float*)(ws + OFF_C1);
  float*    c2    = (float*)(ws + OFF_C2);
  __bf16*   h1bf  = (__bf16*)(ws + OFF_H1BF);
  __bf16*   h2bf  = (__bf16*)(ws + OFF_H2BF);
  float*    obuf  = (float*)(ws + OFF_OBUF);
  unsigned* bar   = (unsigned*)(ws + OFF_BAR);
  __bf16*   w1    = (__bf16*)(ws + OFF_W1);
  __bf16*   w2    = (__bf16*)(ws + OFF_W2);
  __bf16*   w3    = (__bf16*)(ws + OFF_W3);
  float*    wih1f = (float*)(ws + OFF_WIH1);
  float*    b1c   = (float*)(ws + OFF_B1);
  float*    b2c   = (float*)(ws + OFF_B2);
  float*    wl    = (float*)(ws + OFF_WLIN);

  // reset recurrent state + barrier every call (graph-replay determinism)
  hipMemsetAsync(ws, 0, ZERO_BYTES, stream);

  prep_weights<<<(4 * H * H) / 256, 256, 0, stream>>>(Whh1, Wih2, Whh2, w1, w2, w3);
  prep_small<<<(4 * H) / 256, 256, 0, stream>>>(Wih1, bih1, bhh1, bih2, bhh2, Wlin, blin,
                                                wih1f, b1c, b2c, wl);

  lstm_persist<<<NBLOCKS, NTHREADS, 0, stream>>>(
      input, w1, w2, w3, wih1f, b1c, b2c, wl,
      c1, c2, h1bf, h2bf, obuf, bar, (float*)d_out);
}
